// NeRF_64922725646359
// MI455X (gfx1250) — compile-verified, matrix operand order checked by reading, not run on hardware
//
#include <hip/hip_runtime.h>
#include <hip/hip_bf16.h>
#include <math.h>

// ---------------------------------------------------------------------------
// Types
// ---------------------------------------------------------------------------
typedef _Float16 v16h __attribute__((ext_vector_type(16)));
typedef _Float16 v8h  __attribute__((ext_vector_type(8)));
typedef float    v8f  __attribute__((ext_vector_type(8)));
typedef int      v4i  __attribute__((ext_vector_type(4)));

#define N_RAYS   2048
#define NC       64
#define NF       128
#define NTOT     192           // NC + NF
#define T_FAR    2.5f
#define MTILE    128           // points per workgroup in the MLP megakernel

// f16-packed weight layout (halves), per parameter set
#define OFF_W0   0             // 64  x 256
#define OFF_W1   16384         // 256 x 256
#define OFF_W2   81920
#define OFF_W3   147456
#define OFF_W4   212992        // 320 x 256 (316 real)
#define OFF_W5   294912
#define OFF_W6   360448
#define OFF_W7   425984
#define OFF_WF   491520        // 256 x 256
#define OFF_WR1  557056        // 288 x 128 (280 real)
#define WSET_HALVES 593920

// Dynamic LDS partition (halves):
//  sH0: 128*256 = 32768   sH1: 32768   sX: 8192   sE: 4096
//  sWB0: 32768 (4 K-chunks x 32 x 256)   sWB1: 32768
#define SMEM_BYTES ((32768 + 32768 + 8192 + 4096 + 32768 + 32768) * 2)

// ---------------------------------------------------------------------------
// Async global->LDS staging (gfx1250 GLOBAL_LOAD_ASYNC_TO_LDS_B128, ASYNCcnt)
// ROCm 7.2 builtin signature takes generic int4* pointers (probe-confirmed by
// the round-2 diagnostic). Guarded: falls back to sync VGPR copy.
// ---------------------------------------------------------------------------
#if defined(__gfx1250__) && __has_builtin(__builtin_amdgcn_global_load_async_to_lds_b128)
#define ASYNC_STAGE 1
#endif

__device__ __forceinline__ void stage_group(_Float16* dst, const _Float16* src,
                                            int halves, int tid) {
#ifdef ASYNC_STAGE
  const v4i* s = (const v4i*)src;
  v4i*       d = (v4i*)dst;
  const int n16 = halves >> 3;  // 16B packets
  for (int i = tid; i < n16; i += 256) {
    __builtin_amdgcn_global_load_async_to_lds_b128((v4i*)(s + i), (v4i*)(d + i),
                                                   0, 0);
  }
#else
  const v8h* s8 = (const v8h*)src;
  v8h*       d8 = (v8h*)dst;
  for (int i = tid; i < halves / 8; i += 256) d8[i] = s8[i];
#endif
}

__device__ __forceinline__ void wait_stage() {
#ifdef ASYNC_STAGE
#if __has_builtin(__builtin_amdgcn_s_wait_asynccnt)
  __builtin_amdgcn_s_wait_asynccnt(0);
#else
  asm volatile("s_wait_asynccnt 0x0" ::: "memory");
#endif
#endif
}

// ---------------------------------------------------------------------------
// WMMA helpers (gfx1250, wave32)
// ---------------------------------------------------------------------------
__device__ __forceinline__ v8f wmma_f16(v16h a, v16h b, v8f c) {
  return __builtin_amdgcn_wmma_f32_16x16x32_f16(false, a, false, b, (short)0, c,
                                                false, false);
}

__device__ __forceinline__ v8f splat8(float x) {
  return (v8f){x, x, x, x, x, x, x, x};
}

// A fragment: 16x32 f16 tile, rows [mrow..mrow+15] of row-major A (lda halves).
// ISA layout: lane group (lane>>4) selects K sub-bytes {half*8..+7, 16+half*8..+7}.
__device__ __forceinline__ v16h load_a(const _Float16* A, int lda, int mrow,
                                       int kc, int lane) {
  const int idx  = lane & 15;
  const int half = (lane >> 4) & 1;
  const _Float16* p = A + (size_t)(mrow + idx) * lda + kc * 32 + half * 8;
  v8h lo = *(const v8h*)p;
  v8h hi = *(const v8h*)(p + 16);
  v16h r;
#pragma unroll
  for (int e = 0; e < 8; ++e) { r[e] = lo[e]; r[8 + e] = hi[e]; }
  return r;
}

// B fragment from pre-swizzled weight staging: contiguous 32B per lane.
// Layout index: ((kc*N + n)*2 + half)*16 + epos
__device__ __forceinline__ v16h load_b(const _Float16* Bst, int N, int ct,
                                       int kc, int lane) {
  const int idx  = lane & 15;
  const int half = (lane >> 4) & 1;
  return *(const v16h*)(Bst + (size_t)(((kc * N + ct * 16 + idx) * 2 + half)) * 16);
}

// C/D layout: VGPR r -> lanes 0-15: (M=r, N=lane), lanes 16-31: (M=8+r, N=lane-16)
__device__ __forceinline__ void store_tile(_Float16* O, int ldo, int mrow,
                                           int ct, int lane, v8f acc, bool relu) {
  const int idx = lane & 15;
  const int mo  = (lane >> 4) ? 8 : 0;
#pragma unroll
  for (int r = 0; r < 8; ++r) {
    float v = acc[r];
    if (relu) v = fmaxf(v, 0.f);
    O[(size_t)(mrow + mo + r) * ldo + ct * 16 + idx] = (_Float16)v;
  }
}

// MAC one staged K-group (NKC chunks of 32) into all NCT column tiles.
// A rows come from segment A1 (chunks [0,n1)) then A2 (chunks [n1,...)).
template <int NCT, int NKC>
__device__ __forceinline__ void group_mac(v8f* acc, const _Float16* A1, int lda1,
                                          int n1, const _Float16* A2, int lda2,
                                          int c0, const _Float16* buf, int mrow,
                                          int lane) {
  v16h af[NKC];
#pragma unroll
  for (int k = 0; k < NKC; ++k) {
    const int c = c0 + k;
    af[k] = (c < n1) ? load_a(A1, lda1, mrow, c, lane)
                     : load_a(A2, lda2, mrow, c - n1, lane);
  }
#pragma unroll
  for (int ct = 0; ct < NCT; ++ct) {
#pragma unroll
    for (int k = 0; k < NKC; ++k)
      acc[ct] = wmma_f16(af[k], load_b(buf, NCT * 16, ct, k, lane), acc[ct]);
  }
}

// One layer Out[128,N] = act(A @ W + bias), W streamed from packed global in
// double-buffered 128-K groups overlapped (async) with WMMA compute.
template <int NCT, int TOTC>
__device__ __attribute__((noinline)) void layer_pipe(
    const _Float16* A1, int lda1, int n1, const _Float16* A2, int lda2,
    const _Float16* __restrict__ wsrc, _Float16* wb0, _Float16* wb1,
    const float* __restrict__ bias, _Float16* out, bool relu, int mrow,
    int lane, int tid) {
  constexpr int N = NCT * 16;
  constexpr int GROUPS = (TOTC + 3) / 4;

  __syncthreads();  // prior layer's reads/writes complete before restaging
  {
    const int nk0 = TOTC >= 4 ? 4 : TOTC;
    stage_group(wb0, wsrc, nk0 * N * 32, tid);
  }
  wait_stage();
  __syncthreads();

  v8f acc[NCT];
  const int bidx = lane & 15;
#pragma unroll
  for (int ct = 0; ct < NCT; ++ct) acc[ct] = splat8(bias[ct * 16 + bidx]);

  _Float16* bufs[2] = {wb0, wb1};
#pragma unroll
  for (int g = 0; g < GROUPS; ++g) {
    const int c0 = g * 4;
    const int nk = (TOTC - c0) >= 4 ? 4 : (TOTC - c0);
    if (g + 1 < GROUPS) {  // prefetch next group while computing this one
      const int c1 = c0 + 4;
      const int nk1 = (TOTC - c1) >= 4 ? 4 : (TOTC - c1);
      stage_group(bufs[(g + 1) & 1], wsrc + (size_t)c1 * N * 32, nk1 * N * 32,
                  tid);
    }
    switch (nk) {
      case 4: group_mac<NCT, 4>(acc, A1, lda1, n1, A2, lda2, c0, bufs[g & 1],
                                mrow, lane); break;
      case 2: group_mac<NCT, 2>(acc, A1, lda1, n1, A2, lda2, c0, bufs[g & 1],
                                mrow, lane); break;
      default: group_mac<NCT, 1>(acc, A1, lda1, n1, A2, lda2, c0, bufs[g & 1],
                                 mrow, lane); break;
    }
    if (g + 1 < GROUPS) {
      wait_stage();
      __syncthreads();
    }
  }
#pragma unroll
  for (int ct = 0; ct < NCT; ++ct)
    store_tile(out, 256, mrow, ct, lane, acc[ct], relu);
}

// ---------------------------------------------------------------------------
// Kernel: convert f32 weights -> f16, pre-swizzled into B-fragment layout
// ---------------------------------------------------------------------------
__global__ void pack_weights(const float* __restrict__ src,
                             _Float16* __restrict__ dst,
                             int Kreal, int Kpad, int N) {
  int idx = blockIdx.x * blockDim.x + threadIdx.x;
  if (idx >= Kpad * N) return;
  int k = idx / N, n = idx % N;
  float v = (k < Kreal) ? src[(size_t)k * N + n] : 0.f;
  int kc = k >> 5, r = k & 31;
  int sub = r >> 4, half = (r >> 3) & 1, e = r & 7;
  dst[(size_t)((kc * N + n) * 2 + half) * 16 + sub * 8 + e] = (_Float16)v;
}

// ---------------------------------------------------------------------------
// Kernel: stratified coarse samples  t_c = (T_FAR/NC) * (s + u_c)
// ---------------------------------------------------------------------------
__global__ void gen_tc(const float* __restrict__ u_c, float* __restrict__ t_c,
                       int n) {
  int i = blockIdx.x * blockDim.x + threadIdx.x;
  if (i >= n) return;
  int s = i & (NC - 1);
  t_c[i] = (T_FAR / NC) * ((float)s + u_c[i]);
}

// ---------------------------------------------------------------------------
// Megakernel: NeRF radiance-field MLP for a tile of 128 points (WMMA chain)
// ---------------------------------------------------------------------------
struct MlpScalars {
  const float* bias[10];   // b0..b7, bf(8), br1(9)
  const float* Ws;
  const float* bs;
  const float* Wr2;
  const float* br2;
};

__global__ __launch_bounds__(256) void nerf_mlp(
    const float* __restrict__ o, const float* __restrict__ dir,
    const float* __restrict__ tbuf, int nsamp,
    const _Float16* __restrict__ wpk, MlpScalars P,
    float* __restrict__ sigmaOut, float* __restrict__ rgbOut) {
  extern __shared__ char smem[];
  _Float16* sH0 = (_Float16*)smem;     // 128x256
  _Float16* sH1 = sH0 + 32768;         // 128x256
  _Float16* sX  = sH1 + 32768;         // 128x64  (gamma(x), 60 real)
  _Float16* sE  = sX + 8192;           // 128x32  (gamma(d), 24 real)
  _Float16* sW0 = sE + 4096;           // K-group ping
  _Float16* sW1 = sW0 + 32768;         // K-group pong

  const int tid  = threadIdx.x;
  const int lane = tid & 31;
  const int wave = tid >> 5;
  const int mrow = wave * 16;
  const int ptBase = blockIdx.x * MTILE;

  // ---- positional encodings (computed in-register, written to LDS) ----
  if (tid < MTILE) {
    const int pt  = ptBase + tid;
    const int ray = pt / nsamp;
    const float t = tbuf[pt];
    float px[3], pd[3];
#pragma unroll
    for (int c = 0; c < 3; ++c) {
      pd[c] = dir[ray * 3 + c];
      px[c] = o[ray * 3 + c] + t * pd[c];
    }
#pragma unroll
    for (int c = 0; c < 3; ++c) {
      float f = 3.14159265358979f;
#pragma unroll
      for (int l = 0; l < 10; ++l) {
        float ang = px[c] * f;
        sX[tid * 64 + c * 20 + l]      = (_Float16)__sinf(ang);
        sX[tid * 64 + c * 20 + 10 + l] = (_Float16)__cosf(ang);
        f *= 2.f;
      }
    }
#pragma unroll
    for (int k = 60; k < 64; ++k) sX[tid * 64 + k] = (_Float16)0.f;
#pragma unroll
    for (int c = 0; c < 3; ++c) {
      float f = 3.14159265358979f;
#pragma unroll
      for (int l = 0; l < 4; ++l) {
        float ang = pd[c] * f;
        sE[tid * 32 + c * 8 + l]     = (_Float16)__sinf(ang);
        sE[tid * 32 + c * 8 + 4 + l] = (_Float16)__cosf(ang);
        f *= 2.f;
      }
    }
#pragma unroll
    for (int k = 24; k < 32; ++k) sE[tid * 32 + k] = (_Float16)0.f;
  }

  _Float16* cur = sH0;
  _Float16* nxt = sH1;

  // ---- layer 0: gamma(x)[64] @ W0 -> 256 ----
  layer_pipe<16, 2>(sX, 64, 2, nullptr, 0, wpk + OFF_W0, sW0, sW1, P.bias[0],
                    cur, true, mrow, lane, tid);

  // ---- layers 1..7 (skip concat [h, gamma(x)] at layer 4) ----
  const int OFFS[8] = {OFF_W0, OFF_W1, OFF_W2, OFF_W3,
                       OFF_W4, OFF_W5, OFF_W6, OFF_W7};
#pragma unroll 1
  for (int L = 1; L < 8; ++L) {
    if (L == 4)
      layer_pipe<16, 10>(cur, 256, 8, sX, 64, wpk + OFF_W4, sW0, sW1,
                         P.bias[4], nxt, true, mrow, lane, tid);
    else
      layer_pipe<16, 8>(cur, 256, 8, nullptr, 0, wpk + OFFS[L], sW0, sW1,
                        P.bias[L], nxt, true, mrow, lane, tid);
    _Float16* tmp = cur; cur = nxt; nxt = tmp;
  }
  __syncthreads();

  // ---- sigma head (N=1): VALU dot ----
  if (tid < MTILE) {
    float s = P.bs[0];
    const _Float16* h = cur + (size_t)tid * 256;
    for (int k = 0; k < 256; ++k) s += (float)h[k] * P.Ws[k];
    sigmaOut[ptBase + tid] = fmaxf(s, 0.f);
  }

  // ---- feature layer (Wf, no relu) ----
  layer_pipe<16, 8>(cur, 256, 8, nullptr, 0, wpk + OFF_WF, sW0, sW1, P.bias[8],
                    nxt, false, mrow, lane, tid);
  { _Float16* tmp = cur; cur = nxt; nxt = tmp; }

  // ---- view branch layer 1: [feat(256), gamma(d)(32)] @ Wr1 -> 128 ----
  layer_pipe<8, 9>(cur, 256, 8, sE, 32, wpk + OFF_WR1, sW0, sW1, P.bias[9],
                   nxt, true, mrow, lane, tid);
  { _Float16* tmp = cur; cur = nxt; nxt = tmp; }
  __syncthreads();

  // ---- view branch layer 2 (N=3) + sigmoid: VALU dots ----
  if (tid < MTILE) {
    const _Float16* h2 = cur + (size_t)tid * 256;   // 128 valid cols
    float r0 = P.br2[0], r1 = P.br2[1], r2 = P.br2[2];
    for (int k = 0; k < 128; ++k) {
      float hv = (float)h2[k];
      r0 += hv * P.Wr2[k * 3 + 0];
      r1 += hv * P.Wr2[k * 3 + 1];
      r2 += hv * P.Wr2[k * 3 + 2];
    }
    const int pt = ptBase + tid;
    rgbOut[pt * 3 + 0] = 1.f / (1.f + __expf(-r0));
    rgbOut[pt * 3 + 1] = 1.f / (1.f + __expf(-r1));
    rgbOut[pt * 3 + 2] = 1.f / (1.f + __expf(-r2));
  }
}

// ---------------------------------------------------------------------------
// Kernel: per-ray volume rendering composite (serial transmittance scan)
// ---------------------------------------------------------------------------
__global__ void composite_k(const float* __restrict__ t,
                            const float* __restrict__ sigma,
                            const float* __restrict__ rgb,
                            float* __restrict__ C, float* __restrict__ wOut,
                            int nsamp, int nrays) {
  int ray = blockIdx.x * blockDim.x + threadIdx.x;
  if (ray >= nrays) return;
  const float* tr = t + (size_t)ray * nsamp;
  const float* sg = sigma + (size_t)ray * nsamp;
  const float* rg = rgb + (size_t)ray * nsamp * 3;
  float T = 1.f, c0 = 0.f, c1 = 0.f, c2 = 0.f, wsum = 0.f;
  for (int s = 0; s < nsamp; ++s) {
    float delta = (s + 1 < nsamp) ? (tr[s + 1] - tr[s]) : 1e8f;
    float em = __expf(-sg[s] * delta);
    float w = T * (1.f - em);
    T *= em;
    c0 += w * rg[s * 3 + 0];
    c1 += w * rg[s * 3 + 1];
    c2 += w * rg[s * 3 + 2];
    wsum += w;
    if (wOut) wOut[(size_t)ray * nsamp + s] = w;
  }
  float bg = 1.f - wsum;  // c_bg = (1,1,1)
  C[ray * 3 + 0] = c0 + bg;
  C[ray * 3 + 1] = c1 + bg;
  C[ray * 3 + 2] = c2 + bg;
}

// ---------------------------------------------------------------------------
// Kernel: inverse-CDF fine sampling + merge-sort (bitonic, 256 padded)
// one block (128 threads) per ray
// ---------------------------------------------------------------------------
__global__ __launch_bounds__(128) void sample_fine_k(
    const float* __restrict__ wC, const float* __restrict__ tC,
    const float* __restrict__ uF, float* __restrict__ tF) {
  __shared__ float cdf[NC + 1];
  __shared__ float arr[256];
  const int ray = blockIdx.x;
  const int tid = threadIdx.x;

  if (tid == 0) {
    float tot = 0.f;
    for (int i = 0; i < NC; ++i) tot += wC[ray * NC + i] + 1e-5f;
    float s = 0.f;
    cdf[0] = 0.f;
    for (int i = 0; i < NC; ++i) {
      s += wC[ray * NC + i] + 1e-5f;
      cdf[i + 1] = s / tot;
    }
  }
  __syncthreads();

  // inverse CDF sample for u_f[tid]
  {
    float u = uF[ray * NF + tid];
    int ind = 0;
    for (int i = 0; i <= NC; ++i) ind += (cdf[i] <= u) ? 1 : 0;  // side='right'
    int below = ind - 1; below = below < 0 ? 0 : (below > NC - 1 ? NC - 1 : below);
    int above = ind;     above = above < 1 ? 1 : (above > NC ? NC : above);
    float c0 = cdf[below], c1 = cdf[above];
    float b0 = (T_FAR / NC) * below, b1 = (T_FAR / NC) * above;
    float den = (c1 - c0 < 1e-5f) ? 1.f : (c1 - c0);
    arr[NC + tid] = b0 + (u - c0) / den * (b1 - b0);   // slots 64..191
  }
  if (tid < NC) arr[tid] = tC[ray * NC + tid];          // slots 0..63
  else arr[128 + tid] = 1e30f;                          // pad slots 192..255
  __syncthreads();

  // bitonic sort of 256 elements with 128 threads
  for (int k = 2; k <= 256; k <<= 1) {
    for (int j = k >> 1; j > 0; j >>= 1) {
      for (int i = tid; i < 256; i += 128) {
        int ixj = i ^ j;
        if (ixj > i) {
          bool up = ((i & k) == 0);
          float a = arr[i], b = arr[ixj];
          if ((a > b) == up) { arr[i] = b; arr[ixj] = a; }
        }
      }
      __syncthreads();
    }
  }
  tF[(size_t)ray * NTOT + tid] = arr[tid];
  if (tid < 64) tF[(size_t)ray * NTOT + 128 + tid] = arr[128 + tid];
}

// ---------------------------------------------------------------------------
// Host launcher
// ---------------------------------------------------------------------------
extern "C" void kernel_launch(void* const* d_in, const int* in_sizes, int n_in,
                              void* d_out, int out_size, void* d_ws,
                              size_t ws_size, hipStream_t stream) {
  (void)in_sizes; (void)n_in; (void)out_size; (void)ws_size;

  const float* o   = (const float*)d_in[0];
  const float* dv  = (const float*)d_in[1];
  const float* u_c = (const float*)d_in[2];
  const float* u_f = (const float*)d_in[3];
  const int PBASE[2] = {4, 28};  // params_c, params_f (24 arrays each: W,b pairs)

  // ---- workspace layout ----
  char* ws = (char*)d_ws;
  _Float16* wpk = (_Float16*)ws;  // both packed f16 weight sets
  size_t cur = (size_t)2 * WSET_HALVES * sizeof(_Float16);
  auto take = [&](size_t bytes) {
    size_t r = cur;
    cur += (bytes + 255) & ~(size_t)255;
    return ws + r;
  };
  float* t_c   = (float*)take((size_t)N_RAYS * NC * 4);
  float* sig_c = (float*)take((size_t)N_RAYS * NC * 4);
  float* rgb_c = (float*)take((size_t)N_RAYS * NC * 12);
  float* w_c   = (float*)take((size_t)N_RAYS * NC * 4);
  float* t_f   = (float*)take((size_t)N_RAYS * NTOT * 4);
  float* sig_f = (float*)take((size_t)N_RAYS * NTOT * 4);
  float* rgb_f = (float*)take((size_t)N_RAYS * NTOT * 12);

  // ---- pack weights (both sets) into f16 B-fragment layout ----
  struct MatDesc { int pidx, Kreal, Kpad, N, off; };
  const MatDesc mats[10] = {
      {0, 60, 64, 256, OFF_W0},   {2, 256, 256, 256, OFF_W1},
      {4, 256, 256, 256, OFF_W2}, {6, 256, 256, 256, OFF_W3},
      {8, 316, 320, 256, OFF_W4}, {10, 256, 256, 256, OFF_W5},
      {12, 256, 256, 256, OFF_W6},{14, 256, 256, 256, OFF_W7},
      {18, 256, 256, 256, OFF_WF},{20, 280, 288, 128, OFF_WR1}};
  for (int set = 0; set < 2; ++set) {
    for (int m = 0; m < 10; ++m) {
      const MatDesc& md = mats[m];
      const float* src = (const float*)d_in[PBASE[set] + md.pidx];
      _Float16* dst = wpk + (size_t)set * WSET_HALVES + md.off;
      int n = md.Kpad * md.N;
      pack_weights<<<(n + 255) / 256, 256, 0, stream>>>(src, dst, md.Kreal,
                                                        md.Kpad, md.N);
    }
  }

  auto make_scalars = [&](int set) {
    MlpScalars P;
    int p0 = PBASE[set];
    for (int i = 0; i < 8; ++i) P.bias[i] = (const float*)d_in[p0 + 2 * i + 1];
    P.bias[8] = (const float*)d_in[p0 + 19];  // bf
    P.bias[9] = (const float*)d_in[p0 + 21];  // br1
    P.Ws  = (const float*)d_in[p0 + 16];
    P.bs  = (const float*)d_in[p0 + 17];
    P.Wr2 = (const float*)d_in[p0 + 22];
    P.br2 = (const float*)d_in[p0 + 23];
    return P;
  };

  // ---- coarse pass ----
  {
    int n = N_RAYS * NC;
    gen_tc<<<(n + 255) / 256, 256, 0, stream>>>(u_c, t_c, n);
    MlpScalars Pc = make_scalars(0);
    int blocks = n / MTILE;
    nerf_mlp<<<blocks, 256, SMEM_BYTES, stream>>>(o, dv, t_c, NC, wpk, Pc,
                                                  sig_c, rgb_c);
    composite_k<<<(N_RAYS + 255) / 256, 256, 0, stream>>>(
        t_c, sig_c, rgb_c, (float*)d_out, w_c, NC, N_RAYS);
  }

  // ---- fine sampling + fine pass ----
  {
    sample_fine_k<<<N_RAYS, 128, 0, stream>>>(w_c, t_c, u_f, t_f);
    MlpScalars Pf = make_scalars(1);
    int n = N_RAYS * NTOT;
    int blocks = n / MTILE;
    nerf_mlp<<<blocks, 256, SMEM_BYTES, stream>>>(
        o, dv, t_f, NTOT, wpk + (size_t)WSET_HALVES, Pf, sig_f, rgb_f);
    composite_k<<<(N_RAYS + 255) / 256, 256, 0, stream>>>(
        t_f, sig_f, rgb_f, (float*)d_out + N_RAYS * 3, nullptr, NTOT, N_RAYS);
  }
}
